// DeepseekMoE_70635032150792
// MI455X (gfx1250) — matrix operand order repaired and checked by reading, not compiled, not run on hardware
//
#include <hip/hip_runtime.h>
#include <hip/hip_bf16.h>

#define H_DIM  2048
#define I_DIM  1408
#define IS_DIM 2816
#define E_NUM  16
#define TOPK   6
#define T_NUM  2048

// GEMM tiling: block = 256 threads = 8 waves (wave32), block tile 128x64,
// wave tile 32x32 (2x2 wmma 16x16x32 bf16), K step 32.
#define BM 128
#define BN 64
#define BK 32
#define LDK 40   // padded LDS row stride (elements), 80B: 16B aligned, bank-spread

typedef __bf16 bf16_t;
typedef __attribute__((ext_vector_type(16))) __bf16 v16bf;
typedef __attribute__((ext_vector_type(8)))  __bf16 v8bf;
typedef __attribute__((ext_vector_type(4)))  __bf16 v4bf;
typedef __attribute__((ext_vector_type(8)))  float  v8f;

// ---- fragment loaders from LDS (row-major [rows][LDK] bf16) -----------------
// A-matrix 16x32 f16 layout (ISA 7.12.2): lane L: row = L%16, h = L/16;
// VGPR0..3 hold K = h*8 .. h*8+7 ; VGPR4..7 hold K = 16+h*8 .. 16+h*8+7
static __device__ inline v16bf frag_a(const bf16_t* lds, int rowBase, int lane) {
  const int r = rowBase + (lane & 15);
  const int h = lane >> 4;
  v8bf lo = *(const v8bf*)(lds + r * LDK + h * 8);
  v8bf hi = *(const v8bf*)(lds + r * LDK + 16 + h * 8);
  v16bf f;
#pragma unroll
  for (int i = 0; i < 8; ++i) { f[i] = lo[i]; f[i + 8] = hi[i]; }
  return f;
}
// B-matrix 32x16 f16 layout: lane L: col = L%16, h = L/16;
// VGPR0..3 hold K = h*8 .. h*8+7 ; VGPR4..7 hold K = 16+h*8 .. 16+h*8+7
static __device__ inline v16bf frag_b(const bf16_t* lds, int rowBase, int lane) {
  const int r = rowBase + (lane & 15);
  const int h = lane >> 4;
  v8bf lo = *(const v8bf*)(lds + r * LDK + h * 8);
  v8bf hi = *(const v8bf*)(lds + r * LDK + 16 + h * 8);
  v16bf f;
#pragma unroll
  for (int i = 0; i < 8; ++i) { f[i] = lo[i]; f[i + 8] = hi[i]; }
  return f;
}

static __device__ inline void pack_store4(bf16_t* d, float4 v) {
  v4bf pk = {(bf16_t)v.x, (bf16_t)v.y, (bf16_t)v.z, (bf16_t)v.w};
  *(v4bf*)d = pk;
}

#define WMMA_BF16(A, B, C) \
  __builtin_amdgcn_wmma_f32_16x16x32_bf16(false, (A), false, (B), (short)0, (C), false, false)

// ---- routing: logits -> softmax -> top-6 -> per-expert (token, weight) lists
__global__ __launch_bounds__(256) void routing_kernel(
    const float* __restrict__ X, const float* __restrict__ GW,
    int* __restrict__ cnt, int* __restrict__ tokIdx, float* __restrict__ tokW) {
  const int wave = threadIdx.x >> 5;
  const int lane = threadIdx.x & 31;
  const int t = blockIdx.x * 8 + wave;  // T = 2048 = gridDim.x(256) * 8

  float acc[E_NUM];
#pragma unroll
  for (int e = 0; e < E_NUM; ++e) acc[e] = 0.f;
  for (int k = lane; k < H_DIM; k += 32) {
    const float xv = X[(size_t)t * H_DIM + k];
#pragma unroll
    for (int e = 0; e < E_NUM; ++e) acc[e] += xv * GW[e * H_DIM + k];
  }
#pragma unroll
  for (int off = 16; off > 0; off >>= 1) {
#pragma unroll
    for (int e = 0; e < E_NUM; ++e) acc[e] += __shfl_xor(acc[e], off, 32);
  }
  if (lane == 0) {
    float m = acc[0];
#pragma unroll
    for (int e = 1; e < E_NUM; ++e) m = fmaxf(m, acc[e]);
    float p[E_NUM]; float s = 0.f;
#pragma unroll
    for (int e = 0; e < E_NUM; ++e) { p[e] = __expf(acc[e] - m); s += p[e]; }
    const float inv = __builtin_amdgcn_rcpf(s);
#pragma unroll
    for (int kk = 0; kk < TOPK; ++kk) {
      int best = 0; float bv = p[0];
#pragma unroll
      for (int e = 1; e < E_NUM; ++e) if (p[e] > bv) { bv = p[e]; best = e; }
      const float w = bv * inv;           // NORM_TOPK_PROB == False
      const int slot = atomicAdd(&cnt[best], 1);
      tokIdx[best * T_NUM + slot] = t;
      tokW[best * T_NUM + slot] = w;
      p[best] = -1.f;
    }
  }
}

__global__ void init_kernel(int* __restrict__ cnt) {
  if (threadIdx.x < E_NUM) cnt[threadIdx.x] = 0;
}

__global__ void scan_kernel(const int* __restrict__ cnt, int* __restrict__ offs) {
  if (threadIdx.x == 0) {
    int s = 0;
    for (int e = 0; e < E_NUM; ++e) { offs[e] = s; s += cnt[e]; }
  }
}

// ---- up projection: act = silu(X@Wg^T) * (X@Wu^T), bf16 out ----------------
// GATHER!=0: per-expert (blockIdx.z), rows gathered via tokIdx, W1 += e*2*I*K.
// Software-pipelined: next tile's global loads overlap current tile's WMMAs.
template <int GATHER>
__global__ __launch_bounds__(256) void up_gemm(
    const float* __restrict__ X, const float* __restrict__ W1,
    bf16_t* __restrict__ act,
    const int* __restrict__ tokIdx, const int* __restrict__ cnt,
    const int* __restrict__ offs, int Icols, int K, int Mfixed) {
  __shared__ alignas(16) bf16_t lA[BM * LDK];
  __shared__ alignas(16) bf16_t lBg[BN * LDK];
  __shared__ alignas(16) bf16_t lBu[BN * LDK];

  const int e = blockIdx.z;
  int M, actRowBase, idxBase;
  const float* Wb;
  if (GATHER) {
    M = cnt[e]; actRowBase = offs[e]; idxBase = e * T_NUM;
    Wb = W1 + (size_t)e * 2 * Icols * K;
  } else {
    M = Mfixed; actRowBase = 0; idxBase = 0; Wb = W1;
  }
  const int mBase = blockIdx.x * BM;
  if (mBase >= M) return;  // (M >= 1 guaranteed here)
  const int nBase = blockIdx.y * BN;

  const int tid = threadIdx.x;
  const int wave = tid >> 5, lane = tid & 31;
  const int wm = wave >> 1, wn = wave & 1;

  // ---- hoisted, branchless staging coordinates (K-invariant) ----
  // A: 128x32 fp32 = 1024 float4 slots -> 4 per thread. For GATHER, rows >= M
  // clamp to M-1: they feed WMMA with data whose outputs are never stored.
  const float* asrc[4]; bf16_t* adst[4];
#pragma unroll
  for (int it = 0; it < 4; ++it) {
    const int i = tid + it * 256;
    const int r = i >> 3, c = (i & 7) << 2;
    int ml = mBase + r;
    if (GATHER && ml >= M) ml = M - 1;
    const int gr = GATHER ? tokIdx[idxBase + ml] : ml;
    asrc[it] = X + (size_t)gr * K + c;
    adst[it] = lA + r * LDK + c;
  }
  // B: 64x32 fp32 per tile (gate + up) = 512 float4 slots -> 2 per thread each.
  const float* bsrcg[2]; const float* bsrcu[2]; bf16_t* bdstg[2]; bf16_t* bdstu[2];
#pragma unroll
  for (int it = 0; it < 2; ++it) {
    const int i = tid + it * 256;
    const int r = i >> 3, c = (i & 7) << 2;
    bsrcg[it] = Wb + (size_t)(nBase + r) * K + c;
    bsrcu[it] = Wb + (size_t)(Icols + nBase + r) * K + c;
    bdstg[it] = lBg + r * LDK + c;
    bdstu[it] = lBu + r * LDK + c;
  }

  const v8f zero = {0.f, 0.f, 0.f, 0.f, 0.f, 0.f, 0.f, 0.f};
  v8f cg[2][2], cu[2][2];
#pragma unroll
  for (int i = 0; i < 2; ++i)
#pragma unroll
    for (int j = 0; j < 2; ++j) { cg[i][j] = zero; cu[i][j] = zero; }

  // prologue: load tile 0 into regs
  float4 va[4], vg[2], vu[2];
#pragma unroll
  for (int it = 0; it < 4; ++it) va[it] = *(const float4*)(asrc[it]);
#pragma unroll
  for (int it = 0; it < 2; ++it) {
    vg[it] = *(const float4*)(bsrcg[it]);
    vu[it] = *(const float4*)(bsrcu[it]);
  }

  for (int k0 = 0; k0 < K; k0 += BK) {
    __syncthreads();   // previous compute done reading LDS
#pragma unroll
    for (int it = 0; it < 4; ++it) pack_store4(adst[it], va[it]);
#pragma unroll
    for (int it = 0; it < 2; ++it) {
      pack_store4(bdstg[it], vg[it]);
      pack_store4(bdstu[it], vu[it]);
    }
    __syncthreads();   // tile k0 visible to all waves

    const int kn = k0 + BK;
    if (kn < K) {      // issue next tile's loads; they overlap the WMMAs below
#pragma unroll
      for (int it = 0; it < 4; ++it) va[it] = *(const float4*)(asrc[it] + kn);
#pragma unroll
      for (int it = 0; it < 2; ++it) {
        vg[it] = *(const float4*)(bsrcg[it] + kn);
        vu[it] = *(const float4*)(bsrcu[it] + kn);
        __builtin_prefetch(bsrcg[it] + kn + BK, 0, 1);
        __builtin_prefetch(bsrcu[it] + kn + BK, 0, 1);
      }
    }

    const v16bf a0 = frag_a(lA, wm * 32, lane);
    const v16bf a1 = frag_a(lA, wm * 32 + 16, lane);
    const v16bf bg0 = frag_b(lBg, wn * 32, lane);
    const v16bf bg1 = frag_b(lBg, wn * 32 + 16, lane);
    const v16bf bu0 = frag_b(lBu, wn * 32, lane);
    const v16bf bu1 = frag_b(lBu, wn * 32 + 16, lane);
    cg[0][0] = WMMA_BF16(a0, bg0, cg[0][0]);
    cg[0][1] = WMMA_BF16(a0, bg1, cg[0][1]);
    cg[1][0] = WMMA_BF16(a1, bg0, cg[1][0]);
    cg[1][1] = WMMA_BF16(a1, bg1, cg[1][1]);
    cu[0][0] = WMMA_BF16(a0, bu0, cu[0][0]);
    cu[0][1] = WMMA_BF16(a0, bu1, cu[0][1]);
    cu[1][0] = WMMA_BF16(a1, bu0, cu[1][0]);
    cu[1][1] = WMMA_BF16(a1, bu1, cu[1][1]);
  }

  // epilogue: fused silu(g)*u = g*u*rcp(1+exp(-g)), bf16 store.
  // C layout: N = lane%16, M = vgpr + 8*(lane/16).
  const int half = lane >> 4, ln = lane & 15;
  const int rBase = mBase + wm * 32 + half * 8;
  bf16_t* obase = act + (size_t)(actRowBase + rBase) * Icols +
                  (nBase + wn * 32 + ln);
#pragma unroll
  for (int tm = 0; tm < 2; ++tm)
#pragma unroll
    for (int g = 0; g < 8; ++g) {
      const int rowL = rBase + tm * 16 + g;
      bf16_t* p = obase + (size_t)(tm * 16 + g) * Icols;
      if (!GATHER || rowL < M) {
#pragma unroll
        for (int tn = 0; tn < 2; ++tn) {
          const float gv = cg[tm][tn][g];
          const float uv = cu[tm][tn][g];
          const float s = gv * uv * __builtin_amdgcn_rcpf(1.f + __expf(-gv));
          p[tn * 16] = (bf16_t)s;
        }
      }
    }
}

// ---- down projection: out = act(bf16) @ W2^T --------------------------------
// MOE: scale by routing weight and atomicAdd into out[token]; else plain store.
// A operand is bf16: double-buffered async global->LDS DMA (ASYNCcnt), so the
// DMA for tile k+1 runs concurrently with the WMMAs on tile k.
template <int MOE>
__global__ __launch_bounds__(256) void down_gemm(
    const bf16_t* __restrict__ act, const float* __restrict__ W2,
    float* __restrict__ out,
    const int* __restrict__ tokIdx, const float* __restrict__ tokW,
    const int* __restrict__ cnt, const int* __restrict__ offs,
    int K, int Mfixed) {
  __shared__ alignas(16) bf16_t lA2[2][BM * LDK];
  __shared__ alignas(16) bf16_t lB[BN * LDK];

  const int e = blockIdx.z;
  int M, aBase, idxBase;
  const float* Wb;
  if (MOE) {
    M = cnt[e]; aBase = offs[e]; idxBase = e * T_NUM;
    Wb = W2 + (size_t)e * H_DIM * K;
  } else {
    M = Mfixed; aBase = 0; idxBase = 0; Wb = W2;
  }
  const int mBase = blockIdx.x * BM;
  if (mBase >= M) return;
  const int nBase = blockIdx.y * BN;

  const int tid = threadIdx.x;
  const int wave = tid >> 5, lane = tid & 31;
  const int wm = wave >> 1, wn = wave & 1;

  // A: 128x32 bf16 = 512 x 16B chunks -> 2 per thread, async DMA to LDS.
  // OOB rows clamped to M-1 (results discarded in epilogue).
  unsigned long long gA[2]; unsigned ldsA[2][2];
#pragma unroll
  for (int it = 0; it < 2; ++it) {
    const int i = tid + it * 256;
    const int r = i >> 2, cc = (i & 3) << 3;
    int ml = mBase + r;
    if (MOE && ml >= M) ml = M - 1;
    gA[it] = (unsigned long long)(act + (size_t)(aBase + ml) * K + cc);
    // flat address low 32 bits == LDS byte offset
    ldsA[0][it] = (unsigned)(size_t)(&lA2[0][0] + r * LDK + cc);
    ldsA[1][it] = (unsigned)(size_t)(&lA2[1][0] + r * LDK + cc);
  }
  // B: 64x32 fp32 -> bf16, 2 float4 per thread.
  const float* bsrc[2]; bf16_t* bdst[2];
#pragma unroll
  for (int it = 0; it < 2; ++it) {
    const int i = tid + it * 256;
    const int r = i >> 3, cc = (i & 7) << 2;
    bsrc[it] = Wb + (size_t)(nBase + r) * K + cc;
    bdst[it] = lB + r * LDK + cc;
  }

  const v8f zero = {0.f, 0.f, 0.f, 0.f, 0.f, 0.f, 0.f, 0.f};
  v8f c[2][2];
#pragma unroll
  for (int i = 0; i < 2; ++i)
#pragma unroll
    for (int j = 0; j < 2; ++j) c[i][j] = zero;

  // prologue: async DMA of A tile 0 into buffer 0; B tile 0 into regs
#pragma unroll
  for (int it = 0; it < 2; ++it) {
    asm volatile("global_load_async_to_lds_b128 %0, %1, off"
                 :: "v"(ldsA[0][it]), "v"(gA[it]) : "memory");
  }
  float4 vb[2];
#pragma unroll
  for (int it = 0; it < 2; ++it) vb[it] = *(const float4*)(bsrc[it]);

  int buf = 0;
  for (int k0 = 0; k0 < K; k0 += BK) {
    __syncthreads();   // prior compute done reading lB and lA2[buf^1]
#pragma unroll
    for (int it = 0; it < 2; ++it) pack_store4(bdst[it], vb[it]);

    const int kn = k0 + BK;
    if (kn < K) {
      // DMA tile k+1 into the other buffer; overlaps WMMAs on tile k
#pragma unroll
      for (int it = 0; it < 2; ++it) {
        unsigned long long ga = gA[it] + (unsigned long long)kn * 2u;
        asm volatile("global_load_async_to_lds_b128 %0, %1, off"
                     :: "v"(ldsA[buf ^ 1][it]), "v"(ga) : "memory");
      }
#pragma unroll
      for (int it = 0; it < 2; ++it) {
        vb[it] = *(const float4*)(bsrc[it] + kn);
        __builtin_prefetch(bsrc[it] + kn + BK, 0, 1);
      }
      // async loads complete in order: wait for tile k's pair only
      asm volatile("s_wait_asynccnt 0x2" ::: "memory");
    } else {
      asm volatile("s_wait_asynccnt 0x0" ::: "memory");
    }
    __syncthreads();   // tile k0 (A via DMA, B via ds_store) visible

    const bf16_t* la = &lA2[buf][0];
    const v16bf a0 = frag_a(la, wm * 32, lane);
    const v16bf a1 = frag_a(la, wm * 32 + 16, lane);
    const v16bf b0 = frag_b(lB, wn * 32, lane);
    const v16bf b1 = frag_b(lB, wn * 32 + 16, lane);
    c[0][0] = WMMA_BF16(a0, b0, c[0][0]);
    c[0][1] = WMMA_BF16(a0, b1, c[0][1]);
    c[1][0] = WMMA_BF16(a1, b0, c[1][0]);
    c[1][1] = WMMA_BF16(a1, b1, c[1][1]);
    buf ^= 1;
  }

  const int half = lane >> 4, ln = lane & 15;
  const int rBase = mBase + wm * 32 + half * 8;
  const int col0 = nBase + wn * 32 + ln;
#pragma unroll
  for (int tm = 0; tm < 2; ++tm)
#pragma unroll
    for (int g = 0; g < 8; ++g) {
      const int rowL = rBase + tm * 16 + g;
      if (MOE) {
        if (rowL < M) {
          const int t = tokIdx[idxBase + rowL];
          const float w = tokW[idxBase + rowL];
          float* p = out + (size_t)t * H_DIM + col0;
#pragma unroll
          for (int tn = 0; tn < 2; ++tn) atomicAdd(p + tn * 16, c[tm][tn][g] * w);
        }
      } else {
        float* p = out + (size_t)rowL * H_DIM + col0;
#pragma unroll
        for (int tn = 0; tn < 2; ++tn) p[tn * 16] = c[tm][tn][g];
      }
    }
}

// ---- workspace layout (bytes) ----------------------------------------------
//  cnt[16]            @ 0
//  offs[16]           @ 64
//  tokIdx[E*T]        @ 128        (131072)
//  tokW[E*T]          @ 131200     (131072)
//  act_shared[T*IS]b  @ 262272     (11534336)  bf16
//  act_moe[6T*I]      @ 11796608   (34603008)  bf16
//  total ~46.4 MB

extern "C" void kernel_launch(void* const* d_in, const int* in_sizes, int n_in,
                              void* d_out, int out_size, void* d_ws, size_t ws_size,
                              hipStream_t stream) {
  (void)in_sizes; (void)n_in; (void)out_size; (void)ws_size;
  const float* x   = (const float*)d_in[0];
  const float* gw  = (const float*)d_in[1];
  const float* w1  = (const float*)d_in[2];
  const float* w2  = (const float*)d_in[3];
  const float* sw1 = (const float*)d_in[4];
  const float* sw2 = (const float*)d_in[5];
  float* out = (float*)d_out;

  char* ws = (char*)d_ws;
  int*    cnt    = (int*)(ws + 0);
  int*    offs   = (int*)(ws + 64);
  int*    tokIdx = (int*)(ws + 128);
  float*  tokW   = (float*)(ws + 131200);
  bf16_t* actSh  = (bf16_t*)(ws + 262272);
  bf16_t* actMoe = (bf16_t*)(ws + 11796608);

  init_kernel<<<1, 32, 0, stream>>>(cnt);
  routing_kernel<<<T_NUM / 8, 256, 0, stream>>>(x, gw, cnt, tokIdx, tokW);
  scan_kernel<<<1, 32, 0, stream>>>(cnt, offs);

  // shared up: M=2048, N=IS=2816 (44 tiles), K=H
  up_gemm<0><<<dim3(T_NUM / BM, IS_DIM / BN, 1), 256, 0, stream>>>(
      x, sw1, actSh, nullptr, nullptr, nullptr, IS_DIM, H_DIM, T_NUM);
  // moe up: per expert, gathered rows, N=I=1408 (22 tiles), K=H
  up_gemm<1><<<dim3(T_NUM / BM, I_DIM / BN, E_NUM), 256, 0, stream>>>(
      x, w1, actMoe, tokIdx, cnt, offs, I_DIM, H_DIM, 0);
  // shared down: out = actSh @ sw2^T  (plain stores initialize out)
  down_gemm<0><<<dim3(T_NUM / BM, H_DIM / BN, 1), 256, 0, stream>>>(
      actSh, sw2, out, nullptr, nullptr, nullptr, nullptr, IS_DIM, T_NUM);
  // moe down: out += w * (actMoe @ w2[e]^T), scattered via atomics
  down_gemm<1><<<dim3(T_NUM / BM, H_DIM / BN, E_NUM), 256, 0, stream>>>(
      actMoe, w2, out, tokIdx, tokW, cnt, offs, I_DIM, 0);
}
